// LoRAMoEModule_1864015806563
// MI455X (gfx1250) — matrix operand order repaired and checked by reading, not compiled, not run on hardware
//
#include <hip/hip_runtime.h>
#include <hip/hip_bf16.h>
#include <stdint.h>

// ---------------------------------------------------------------------------
// LoRA-MoE fused module for MI455X (gfx1250, wave32, WMMA)
//   out = x@W^T + b + SCALE*(x@bd^T)@bu^T + tw*SCALE*sum_e g_e (x@ed_e^T)@eu_e^T
// Strategy: bf16x3 split-precision WMMA GEMM with the rank-20 LoRA/MoE part
// folded in as 32 extra K columns (K = 3072 + 32 = 3104).
// GEMM staging: double-buffered GLOBAL_LOAD_ASYNC_TO_LDS (ASYNCcnt) when the
// builtins exist; VGPR-staged sync copy fallback otherwise.
// ---------------------------------------------------------------------------

typedef __attribute__((ext_vector_type(16))) __bf16 v16bf;
typedef __attribute__((ext_vector_type(8)))  float  v8f;
typedef int v4i_vs __attribute__((vector_size(16)));   // matches async builtin param type

#define D_IN   3072
#define D_OUT  3072
#define M_TOK  8192          // B*S = 2*4096
#define KEXT   3104          // 3072 + 32 (20 ranks + 12 zero pad)
#define NRANK  20            // 4 base + 16 expert (E=4 * R=4)
#define SCALE_F 0.25f
#define BM 128
#define BN 128
#define LDT 40               // padded LDS row stride (ushorts): bank-stride 20 -> conflict free
#define KSTEPS (KEXT / 32)   // 97

#define AS1 __attribute__((address_space(1)))
#define AS3 __attribute__((address_space(3)))

#if __has_builtin(__builtin_amdgcn_global_load_async_to_lds_b128)
#define HAVE_ASYNC_LDS 1
#else
#define HAVE_ASYNC_LDS 0
#endif

__device__ __forceinline__ void wait_async0() {
#if __has_builtin(__builtin_amdgcn_s_wait_asynccnt)
    __builtin_amdgcn_s_wait_asynccnt(0);
#else
    asm volatile("s_wait_asynccnt 0" ::: "memory");
#endif
}

#if HAVE_ASYNC_LDS
__device__ __forceinline__ void async_ld_b128(const unsigned short* g, unsigned short* l) {
    __builtin_amdgcn_global_load_async_to_lds_b128(
        (AS1 v4i_vs*)g, (AS3 v4i_vs*)l, 0, 0);
}
#endif

__device__ __forceinline__ unsigned short f2bf(float f) {
    unsigned u = __builtin_bit_cast(unsigned, f);
    unsigned r = u + 0x7FFFu + ((u >> 16) & 1u);   // round-to-nearest-even
    return (unsigned short)(r >> 16);
}
__device__ __forceinline__ float bf2f(unsigned short h) {
    unsigned u = ((unsigned)h) << 16;
    return __builtin_bit_cast(float, u);
}

// ---------------- timestep gate: tw = softmax(t*w+b) . [0, 0.5, 1] ----------
__global__ void tw_kernel(const float* __restrict__ tgw, const float* __restrict__ tgb,
                          const float* __restrict__ ts, float* __restrict__ twp) {
    if (threadIdx.x == 0) {
        float t = ts[0];
        float l0 = fmaf(t, tgw[0], tgb[0]);
        float l1 = fmaf(t, tgw[1], tgb[1]);
        float l2 = fmaf(t, tgw[2], tgb[2]);
        float m = fmaxf(l0, fmaxf(l1, l2));
        float e0 = __expf(l0 - m), e1 = __expf(l1 - m), e2 = __expf(l2 - m);
        twp[0] = (0.5f * e1 + 1.0f * e2) / (e0 + e1 + e2);
    }
}

// ------------- fp32 [rows, 3072] -> bf16 hi/lo [rows, 3104] (cols 0..3071) --
__global__ void cvt_hilo_kernel(const float* __restrict__ src,
                                unsigned short* __restrict__ hi,
                                unsigned short* __restrict__ lo) {
    long long row = blockIdx.y;
    int c = (blockIdx.x * blockDim.x + threadIdx.x) * 4;   // grid.x*256*4 == 3072 exactly
    float4 v = *reinterpret_cast<const float4*>(src + row * D_IN + c);
    float f[4] = {v.x, v.y, v.z, v.w};
    unsigned short hs[4], ls[4];
#pragma unroll
    for (int i = 0; i < 4; ++i) {
        hs[i] = f2bf(f[i]);
        ls[i] = f2bf(f[i] - bf2f(hs[i]));
    }
    long long o = row * KEXT + c;
    *reinterpret_cast<ushort4*>(&hi[o]) = make_ushort4(hs[0], hs[1], hs[2], hs[3]);
    *reinterpret_cast<ushort4*>(&lo[o]) = make_ushort4(ls[0], ls[1], ls[2], ls[3]);
}

// ------------- per-token rank coefficients -> A extension columns ----------
__global__ __launch_bounds__(256)
void coef_kernel(const float* __restrict__ x, const float* __restrict__ base_down,
                 const float* __restrict__ expert_down, const float* __restrict__ gates,
                 const float* __restrict__ twp,
                 unsigned short* __restrict__ Ahi, unsigned short* __restrict__ Alo) {
    const int tok = blockIdx.x;
    const int tid = threadIdx.x;
    const float* xp = x + (long long)tok * D_IN;
    float acc[NRANK];
#pragma unroll
    for (int j = 0; j < NRANK; ++j) acc[j] = 0.f;

    for (int i = tid; i < D_IN; i += 256) {
        float xv = xp[i];
#pragma unroll
        for (int r = 0; r < 4; ++r)  acc[r]     = fmaf(xv, base_down[r * D_IN + i], acc[r]);
#pragma unroll
        for (int j = 0; j < 16; ++j) acc[4 + j] = fmaf(xv, expert_down[j * D_IN + i], acc[4 + j]);
    }
    // wave32 reduce
#pragma unroll
    for (int j = 0; j < NRANK; ++j)
#pragma unroll
        for (int off = 16; off > 0; off >>= 1)
            acc[j] += __shfl_xor(acc[j], off, 32);

    __shared__ float red[8][NRANK];
    const int wid = tid >> 5, lane = tid & 31;
    if (lane == 0)
#pragma unroll
        for (int j = 0; j < NRANK; ++j) red[wid][j] = acc[j];
    __syncthreads();

    if (tid < 32) {   // threads 0..19 -> real ranks, 20..31 -> zero padding
        float c = 0.f;
        if (tid < NRANK) {
#pragma unroll
            for (int w = 0; w < 8; ++w) c += red[w][tid];
            if (tid < 4) c *= SCALE_F;
            else {
                int e = (tid - 4) >> 2;
                c *= SCALE_F * twp[0] * gates[e];
            }
        }
        unsigned short h = f2bf(c);
        unsigned short l = f2bf(c - bf2f(h));
        long long o = (long long)tok * KEXT + D_IN + tid;
        Ahi[o] = h; Alo[o] = l;
    }
}

// ------------- pack base_up / expert_up -> B extension columns -------------
__global__ void packU_kernel(const float* __restrict__ base_up, const float* __restrict__ expert_up,
                             unsigned short* __restrict__ Bhi, unsigned short* __restrict__ Blo) {
    int t = blockIdx.x * blockDim.x + threadIdx.x;
    if (t >= D_OUT * 32) return;
    int o = t >> 5, j = t & 31;
    float v = 0.f;
    if (j < 4) v = base_up[o * 4 + j];
    else if (j < NRANK) {
        int e = (j - 4) >> 2, r = (j - 4) & 3;
        v = expert_up[((long long)e * D_OUT + o) * 4 + r];
    }
    unsigned short h = f2bf(v);
    unsigned short l = f2bf(v - bf2f(h));
    long long off = (long long)o * KEXT + D_IN + j;
    Bhi[off] = h; Blo[off] = l;
}

// ------------- main bf16x3 WMMA GEMM: out = A(hi+lo) @ B(hi+lo)^T + bias ----
__global__ __launch_bounds__(256)
void gemm_kernel(const unsigned short* __restrict__ Ahi, const unsigned short* __restrict__ Alo,
                 const unsigned short* __restrict__ Bhi, const unsigned short* __restrict__ Blo,
                 const float* __restrict__ bias, float* __restrict__ out) {
#if HAVE_ASYNC_LDS
    __shared__ __align__(16) unsigned short lds[2 * 4 * 128 * LDT];   // double buffered
#else
    __shared__ __align__(16) unsigned short lds[4 * 128 * LDT];
#endif

    const int tid  = threadIdx.x;
    const int wid  = tid >> 5;        // 0..7 waves
    const int lane = tid & 31;
    const int wm   = wid >> 1;        // 0..3 -> 32-row strip
    const int wn   = wid & 1;         // 0..1 -> 64-col strip
    const int half = lane >> 4;
    const int l15  = lane & 15;

    const long long blkM = (long long)blockIdx.y * BM;
    const long long blkN = (long long)blockIdx.x * BN;

    v8f C[2][4];
#pragma unroll
    for (int i = 0; i < 2; ++i)
#pragma unroll
        for (int j = 0; j < 4; ++j)
#pragma unroll
            for (int e = 0; e < 8; ++e) C[i][j][e] = 0.f;

    const int crow = tid >> 2;              // 0..63
    const int cchk = (tid & 3) * 8;         // 16B chunk within 32-col k-tile

    // fragment compute over one 4-tile stage
    auto compute = [&](const unsigned short* sAh, const unsigned short* sAl,
                       const unsigned short* sBh, const unsigned short* sBl) {
        // A fragments: 16-bit A layout -> elems 0..7 K=half*8+j, elems 8..15 K=16+half*8+j
        v16bf Ah[2], Al[2];
#pragma unroll
        for (int fm = 0; fm < 2; ++fm) {
            int row = wm * 32 + fm * 16 + l15;
            int o0 = row * LDT + half * 8;
            reinterpret_cast<uint4*>(&Ah[fm])[0] = *reinterpret_cast<const uint4*>(&sAh[o0]);
            reinterpret_cast<uint4*>(&Ah[fm])[1] = *reinterpret_cast<const uint4*>(&sAh[o0 + 16]);
            reinterpret_cast<uint4*>(&Al[fm])[0] = *reinterpret_cast<const uint4*>(&sAl[o0]);
            reinterpret_cast<uint4*>(&Al[fm])[1] = *reinterpret_cast<const uint4*>(&sAl[o0 + 16]);
        }
#pragma unroll
        for (int fn = 0; fn < 4; ++fn) {
            // B fragments: 16-bit B layout -> elems 0..7 K=half*8+j, elems 8..15 K=16+half*8+j, col = lane&15
            int row = wn * 64 + fn * 16 + l15;
            int o0 = row * LDT + half * 8;
            v16bf Bh, Bl;
            reinterpret_cast<uint4*>(&Bh)[0] = *reinterpret_cast<const uint4*>(&sBh[o0]);
            reinterpret_cast<uint4*>(&Bh)[1] = *reinterpret_cast<const uint4*>(&sBh[o0 + 16]);
            reinterpret_cast<uint4*>(&Bl)[0] = *reinterpret_cast<const uint4*>(&sBl[o0]);
            reinterpret_cast<uint4*>(&Bl)[1] = *reinterpret_cast<const uint4*>(&sBl[o0 + 16]);
#pragma unroll
            for (int fm = 0; fm < 2; ++fm) {
                C[fm][fn] = __builtin_amdgcn_wmma_f32_16x16x32_bf16(false, Ah[fm], false, Bh,
                                                                   (short)0, C[fm][fn], false, false);
                C[fm][fn] = __builtin_amdgcn_wmma_f32_16x16x32_bf16(false, Ah[fm], false, Bl,
                                                                   (short)0, C[fm][fn], false, false);
                C[fm][fn] = __builtin_amdgcn_wmma_f32_16x16x32_bf16(false, Al[fm], false, Bh,
                                                                   (short)0, C[fm][fn], false, false);
            }
        }
    };

#if HAVE_ASYNC_LDS
    // async memory->LDS fill of one stage (8 x b128 per thread, ASYNCcnt-tracked)
    auto issue = [&](int stage, int k0) {
        unsigned short* sAh = lds + (stage * 4 + 0) * 128 * LDT;
        unsigned short* sAl = lds + (stage * 4 + 1) * 128 * LDT;
        unsigned short* sBh = lds + (stage * 4 + 2) * 128 * LDT;
        unsigned short* sBl = lds + (stage * 4 + 3) * 128 * LDT;
#pragma unroll
        for (int rr = 0; rr < 2; ++rr) {
            int r = crow + rr * 64;
            long long ga = (blkM + r) * (long long)KEXT + k0 + cchk;
            long long gb = (blkN + r) * (long long)KEXT + k0 + cchk;
            int lo = r * LDT + cchk;
            async_ld_b128(&Ahi[ga], &sAh[lo]);
            async_ld_b128(&Alo[ga], &sAl[lo]);
            async_ld_b128(&Bhi[gb], &sBh[lo]);
            async_ld_b128(&Blo[gb], &sBl[lo]);
        }
    };

    issue(0, 0);                          // prologue: fill stage 0
    for (int i = 0; i < KSTEPS; ++i) {
        int cur = i & 1;
        wait_async0();                    // this wave's fills for stage cur done
        __syncthreads();                  // every wave's fills for stage cur done
        if (i + 1 < KSTEPS) issue(cur ^ 1, (i + 1) * 32);   // overlap next fill w/ compute
        const unsigned short* base = lds + cur * 4 * 128 * LDT;
        compute(base, base + 128 * LDT, base + 2 * 128 * LDT, base + 3 * 128 * LDT);
        __syncthreads();                  // all reads of stage cur done before it is refilled
    }
#else
    unsigned short* sAh = lds;
    unsigned short* sAl = lds + 1 * 128 * LDT;
    unsigned short* sBh = lds + 2 * 128 * LDT;
    unsigned short* sBl = lds + 3 * 128 * LDT;
    for (int k0 = 0; k0 < KEXT; k0 += 32) {
        __syncthreads();
#pragma unroll
        for (int rr = 0; rr < 2; ++rr) {
            int r = crow + rr * 64;
            long long ga = (blkM + r) * (long long)KEXT + k0 + cchk;
            long long gb = (blkN + r) * (long long)KEXT + k0 + cchk;
            int lo = r * LDT + cchk;
            *reinterpret_cast<uint4*>(&sAh[lo]) = *reinterpret_cast<const uint4*>(&Ahi[ga]);
            *reinterpret_cast<uint4*>(&sAl[lo]) = *reinterpret_cast<const uint4*>(&Alo[ga]);
            *reinterpret_cast<uint4*>(&sBh[lo]) = *reinterpret_cast<const uint4*>(&Bhi[gb]);
            *reinterpret_cast<uint4*>(&sBl[lo]) = *reinterpret_cast<const uint4*>(&Blo[gb]);
        }
        __syncthreads();
        compute(sAh, sAl, sBh, sBl);
    }
#endif

    // Epilogue: C layout -> M = v + half*8, N = lane&15 ; add bias, store fp32
#pragma unroll
    for (int fn = 0; fn < 4; ++fn) {
        long long col = blkN + wn * 64 + fn * 16 + l15;
        float bv = bias[col];
#pragma unroll
        for (int fm = 0; fm < 2; ++fm) {
            long long rowbase = blkM + wm * 32 + fm * 16 + half * 8;
#pragma unroll
            for (int v = 0; v < 8; ++v)
                out[(rowbase + v) * (long long)D_OUT + col] = C[fm][fn][v] + bv;
        }
    }
}

// ---------------------------------------------------------------------------
extern "C" void kernel_launch(void* const* d_in, const int* in_sizes, int n_in,
                              void* d_out, int out_size, void* d_ws, size_t ws_size,
                              hipStream_t stream) {
    const float* x           = (const float*)d_in[0];
    const float* W           = (const float*)d_in[1];
    const float* b           = (const float*)d_in[2];
    const float* base_down   = (const float*)d_in[3];
    const float* base_up     = (const float*)d_in[4];
    const float* expert_down = (const float*)d_in[5];
    const float* expert_up   = (const float*)d_in[6];
    const float* gates       = (const float*)d_in[7];
    const float* tgate_w     = (const float*)d_in[8];
    const float* tgate_b     = (const float*)d_in[9];
    const float* timestep    = (const float*)d_in[10];
    float* out = (float*)d_out;

    // workspace layout
    float* twp = (float*)d_ws;                                   // 256B reserved
    unsigned short* Ahi = (unsigned short*)((char*)d_ws + 256);  // [8192, 3104]
    unsigned short* Alo = Ahi + (size_t)M_TOK * KEXT;
    unsigned short* Bhi = Alo + (size_t)M_TOK * KEXT;            // [3072, 3104]
    unsigned short* Blo = Bhi + (size_t)D_OUT * KEXT;

    tw_kernel<<<1, 32, 0, stream>>>(tgate_w, tgate_b, timestep, twp);

    cvt_hilo_kernel<<<dim3(3, M_TOK), 256, 0, stream>>>(x, Ahi, Alo);
    cvt_hilo_kernel<<<dim3(3, D_OUT), 256, 0, stream>>>(W, Bhi, Blo);

    coef_kernel<<<M_TOK, 256, 0, stream>>>(x, base_down, expert_down, gates, twp, Ahi, Alo);
    packU_kernel<<<(D_OUT * 32 + 255) / 256, 256, 0, stream>>>(base_up, expert_up, Bhi, Blo);

    gemm_kernel<<<dim3(D_OUT / BN, M_TOK / BM), 256, 0, stream>>>(
        Ahi, Alo, Bhi, Blo, b, out);
}
